// VectorQuantizer_28432683499581
// MI455X (gfx1250) — matrix operand order repaired and checked by reading, not compiled, not run on hardware
//
#include <hip/hip_runtime.h>

// ---------------------------------------------------------------------------
// VectorQuantizer for MI455X (gfx1250, wave32).
//   distances = normalize(inputs) @ embeddings^T   (N=32768, K=32, M=8192)
// v2: prep kernel pre-converts the codebook to interleaved bf16 hi/lo planes
// in workspace; main kernel streams 64KB chunks into double-buffered LDS via
// the Tensor Data Mover (tensor_load_to_lds + s_wait_tensorcnt), overlapping
// DMA with the 3-WMMA hi/lo-split bf16 GEMM + fused register argmax.
// Deterministic loss reduction through workspace partials (no atomics).
// ---------------------------------------------------------------------------

typedef __attribute__((ext_vector_type(16))) __bf16 v16bf;
typedef __attribute__((ext_vector_type(8)))  float  v8f;
typedef unsigned int u32;
typedef __attribute__((ext_vector_type(4))) u32 u32x4;
typedef __attribute__((ext_vector_type(8))) int i32x8;
typedef __attribute__((ext_vector_type(4))) int i32x4;

#define N_ROWS      32768
#define DIMV        32
#define N_CODES     8192
#define CHUNK       512                 // codes staged per TDM transfer
#define ROWS_PER_WG 128                 // 8 waves x 16 rows
#define THREADS     256
#define N_WG        (N_ROWS / ROWS_PER_WG)     // 256
#define N_CHUNKS    (N_CODES / CHUNK)          // 16
#define TILES_PER_CHUNK (CHUNK / 16)           // 32

// ws layout: interleaved bf16 codebook row = [32 hi ushorts][32 lo ushorts]
// (128 B per code, 1 MB total), then 256 f32 loss partials.
#define WS_CODE_U32S   (N_CODES * 32)          // 1 MB as uints
#define CHUNK_BYTES    (CHUNK * 128)           // 64 KB
#define CHUNK_U32S     (CHUNK * 32)

union BF16x16 { uint4 q[2]; v16bf v; unsigned short u[16]; };

__device__ __forceinline__ unsigned short f2bf(float x) {   // RNE f32 -> bf16
    unsigned int u = __float_as_uint(x);
    u += 0x7FFFu + ((u >> 16) & 1u);
    return (unsigned short)(u >> 16);
}
__device__ __forceinline__ float bf2f(unsigned short h) {
    return __uint_as_float(((unsigned int)h) << 16);
}

#if __has_builtin(__builtin_amdgcn_tensor_load_to_lds)
#define HAVE_TDM 1
#else
#define HAVE_TDM 0
#endif

__device__ __forceinline__ void wait_tensorcnt0() {
#if __has_builtin(__builtin_amdgcn_s_wait_tensorcnt)
    __builtin_amdgcn_s_wait_tensorcnt(0);
#else
    asm volatile("s_wait_tensorcnt 0x0" ::: "memory");
#endif
}

#if HAVE_TDM
// 1-D TDM copy: nElems8B 8-byte elements, global -> LDS (D# per ISA 8.3-8.6).
__device__ __forceinline__ void tdm_load_1d(u32 lds_off, const void* gptr, u32 nElems8B)
{
    unsigned long long ga = (unsigned long long)gptr;
    u32x4 g0;
    g0[0] = 1u;                                       // count=1, user mode
    g0[1] = lds_off;                                  // lds_addr [63:32]
    g0[2] = (u32)(ga & 0xFFFFFFFFu);                  // global_addr[31:0]
    g0[3] = (u32)((ga >> 32) & 0x1FFFFFFu) | (2u << 30); // addr[56:32] | type=2
    i32x8 g1;
    g1[0] = (int)(3u << 16);                          // data_size=3 (8B)
    g1[1] = (int)((nElems8B & 0xFFFFu) << 16);        // tensor_dim0[15:0]
    g1[2] = (int)(((nElems8B >> 16) & 0xFFFFu) | (1u << 16)); // td0[31:16] | tensor_dim1=1
    g1[3] = (int)((nElems8B & 0xFFFFu) << 16);        // tile_dim0
    g1[4] = (int)(1u | (1u << 16));                   // tile_dim1=1, tile_dim2=1
    g1[5] = (int)nElems8B;                            // tensor_dim0_stride[31:0]
    g1[6] = 0;                                        // stride0[47:32], stride1[15:0]
    g1[7] = 0;                                        // stride1[47:16]
    i32x4 g2;
    g2[0] = 1;                                        // tensor_dim2=1
    g2[1] = 1;                                        // tensor_dim3=1
    g2[2] = 0;                                        // tensor_dim2_stride[31:0]
    g2[3] = (int)(1u << 16);                          // tile_dim3=1
    i32x4 g3;
    g3[0] = 0;                                        // tensor_dim3_stride[31:0]
    g3[1] = (int)(1u << 16);                          // tensor_dim4=1
    g3[2] = (int)(1u << 16);                          // tile_dim4=1
    g3[3] = 0;
#if defined(__clang_major__) && (__clang_major__ >= 23)
    i32x8 g4 = {};
    __builtin_amdgcn_tensor_load_to_lds(g0, g1, g2, g3, g4, 0);
#else
    __builtin_amdgcn_tensor_load_to_lds(g0, g1, g2, g3, 0);
#endif
}
#endif

// ---- prep: codebook f32 -> interleaved bf16 hi/lo rows in workspace -------
__global__ __launch_bounds__(256)
void vq_prep_kernel(const float* __restrict__ codes, unsigned int* __restrict__ wsCodes)
{
    const int i = blockIdx.x * 256 + threadIdx.x;      // code row, 8192 total
    const float4* r4 = (const float4*)(codes + i * DIMV);
    u32 hi[16], lo[16];
    #pragma unroll
    for (int q = 0; q < 8; ++q) {
        float4 v = r4[q];
        unsigned short hx = f2bf(v.x), hy = f2bf(v.y);
        unsigned short hz = f2bf(v.z), hw = f2bf(v.w);
        hi[q * 2 + 0] = (u32)hx | ((u32)hy << 16);
        hi[q * 2 + 1] = (u32)hz | ((u32)hw << 16);
        lo[q * 2 + 0] = (u32)f2bf(v.x - bf2f(hx)) | ((u32)f2bf(v.y - bf2f(hy)) << 16);
        lo[q * 2 + 1] = (u32)f2bf(v.z - bf2f(hz)) | ((u32)f2bf(v.w - bf2f(hw)) << 16);
    }
    uint4* dst = (uint4*)(wsCodes + i * 32);
    #pragma unroll
    for (int q = 0; q < 4; ++q) dst[q]     = ((const uint4*)hi)[q];
    #pragma unroll
    for (int q = 0; q < 4; ++q) dst[4 + q] = ((const uint4*)lo)[q];
}

// ---- main: WMMA similarity + fused argmax ---------------------------------
__global__ __launch_bounds__(THREADS)
void vq_main_kernel(const float* __restrict__ inputs,
                    const float* __restrict__ codes,
                    const unsigned int* __restrict__ wsCodes,
                    float* __restrict__ outQuant,
                    float* __restrict__ outIdx,
                    float* __restrict__ wsPartial)
{
    __shared__ unsigned int sB[2][CHUNK_U32S];   // 2 x 64KB double buffer
    __shared__ int   sIdx[ROWS_PER_WG];
    __shared__ float sLoss[THREADS / 32];

    const int tid  = threadIdx.x;
    const int lane = tid & 31;
    const int wave = tid >> 5;
    const int half = lane >> 4;   // 0: lanes 0-15, 1: lanes 16-31
    const int m    = lane & 15;

    // ---- Normalized A tile (16 rows x K=32) in bf16 hi/lo registers.
    // A in the contiguous k order used here: lanes<16 hold row m, K 0..15;
    // lanes>=16 hold row m, K 16..31; elements K-ascending (matches B).
    const int rowg = blockIdx.x * ROWS_PER_WG + wave * 16 + m;
    const float* arow = inputs + rowg * DIMV;
    float a[16];
    {
        const float4* a4 = (const float4*)arow;
        float4 t0 = a4[half * 4 + 0];
        float4 t1 = a4[half * 4 + 1];
        float4 t2 = a4[half * 4 + 2];
        float4 t3 = a4[half * 4 + 3];
        a[0]=t0.x;  a[1]=t0.y;  a[2]=t0.z;  a[3]=t0.w;
        a[4]=t1.x;  a[5]=t1.y;  a[6]=t1.z;  a[7]=t1.w;
        a[8]=t2.x;  a[9]=t2.y;  a[10]=t2.z; a[11]=t2.w;
        a[12]=t3.x; a[13]=t3.y; a[14]=t3.z; a[15]=t3.w;
    }
    float ss = 0.f;
    #pragma unroll
    for (int i = 0; i < 16; ++i) ss += a[i] * a[i];
    ss += __shfl_xor(ss, 16, 32);          // partner lane holds other 16 elems
    const float scale = 1.0f / fmaxf(sqrtf(ss), 1e-12f);

    BF16x16 ahi, alo;
    #pragma unroll
    for (int i = 0; i < 16; ++i) {
        float x = a[i] * scale;
        unsigned short h = f2bf(x);
        ahi.u[i] = h;
        alo.u[i] = f2bf(x - bf2f(h));
    }

    float best[8];
    int   bidx[8];
    #pragma unroll
    for (int j = 0; j < 8; ++j) { best[j] = -1e30f; bidx[j] = 0; }

#if HAVE_TDM
    const u32 ldsOff[2] = { (u32)(unsigned long long)&sB[0][0],
                            (u32)(unsigned long long)&sB[1][0] };
    const char* wsB = (const char*)wsCodes;
    if (wave == 0)                                  // kick off chunk 0 DMA
        tdm_load_1d(ldsOff[0], wsB, CHUNK_BYTES / 8);
#endif

    for (int c = 0; c < N_CHUNKS; ++c) {
#if HAVE_TDM
        if (wave == 0) wait_tensorcnt0();           // chunk c resident
        __syncthreads();                            // release all waves
        if (wave == 0 && c + 1 < N_CHUNKS)          // overlap next DMA w/ math
            tdm_load_1d(ldsOff[(c + 1) & 1],
                        (const char*)wsCodes + (size_t)(c + 1) * CHUNK_BYTES,
                        CHUNK_BYTES / 8);
#else
        __syncthreads();
        {   // cooperative copy fallback (no TDM builtin on this toolchain)
            const uint4* src = (const uint4*)(wsCodes + (size_t)c * CHUNK_U32S);
            uint4* dst = (uint4*)&sB[c & 1][0];
            #pragma unroll
            for (int it = 0; it < (CHUNK_U32S / 4) / THREADS; ++it)
                dst[it * THREADS + tid] = src[it * THREADS + tid];
        }
        __syncthreads();
#endif
        const unsigned int* bufp = &sB[c & 1][0];

        // 32 tiles of 16 codes: 3 WMMAs each (hi*hi + hi*lo + lo*hi).
        // B in the same contiguous k order: lanes<16 = column N=m K 0..15; lanes>=16 = K 16..31.
        #pragma unroll 4
        for (int t = 0; t < TILES_PER_CHUNK; ++t) {
            const int code_local = t * 16 + m;
            const int ub = code_local * 32 + half * 8;    // uint offset in row
            BF16x16 bhi, blo;
            bhi.q[0] = *(const uint4*)&bufp[ub];
            bhi.q[1] = *(const uint4*)&bufp[ub + 4];
            blo.q[0] = *(const uint4*)&bufp[ub + 16];
            blo.q[1] = *(const uint4*)&bufp[ub + 20];

            v8f acc = {};
            acc = __builtin_amdgcn_wmma_f32_16x16x32_bf16(
                      false, ahi.v, false, bhi.v, (short)0, acc, false, false);
            acc = __builtin_amdgcn_wmma_f32_16x16x32_bf16(
                      false, ahi.v, false, blo.v, (short)0, acc, false, false);
            acc = __builtin_amdgcn_wmma_f32_16x16x32_bf16(
                      false, alo.v, false, bhi.v, (short)0, acc, false, false);

            const int ccode = c * CHUNK + code_local;
            #pragma unroll
            for (int j = 0; j < 8; ++j) {   // ascending codes: '>' keeps first max
                if (acc[j] > best[j]) { best[j] = acc[j]; bidx[j] = ccode; }
            }
        }
    }

    // ---- Cross-lane argmax: rows 0-7 in lanes 0-15, rows 8-15 in 16-31.
    #pragma unroll
    for (int s = 1; s < 16; s <<= 1) {
        #pragma unroll
        for (int j = 0; j < 8; ++j) {
            float ov = __shfl_xor(best[j], s, 32);
            int   oi = __shfl_xor(bidx[j], s, 32);
            if (ov > best[j] || (ov == best[j] && oi < bidx[j])) {
                best[j] = ov; bidx[j] = oi;
            }
        }
    }
    if (m == 0) {                           // lanes 0 and 16 hold winners
        #pragma unroll
        for (int j = 0; j < 8; ++j)
            sIdx[wave * 16 + half * 8 + j] = bidx[j];
    }
    __syncthreads();

    // ---- Gather quantized rows, write indices, accumulate MSE partial.
    float ls = 0.f;
    {
        const int rl   = tid >> 1;
        const int part = tid & 1;
        const int rg   = blockIdx.x * ROWS_PER_WG + rl;
        const int idx  = sIdx[rl];
        if (part == 0) outIdx[rg] = (float)idx;
        const float4* e4 = (const float4*)(codes  + idx * DIMV) + part * 4;
        const float4* i4 = (const float4*)(inputs + rg  * DIMV) + part * 4;
        float4*       o4 = (float4*)(outQuant + rg * DIMV) + part * 4;
        #pragma unroll
        for (int q = 0; q < 4; ++q) {
            float4 e = e4[q], x = i4[q];
            o4[q] = e;                      // straight-through fwd == quantized
            float dx = e.x - x.x, dy = e.y - x.y, dz = e.z - x.z, dw = e.w - x.w;
            ls += dx * dx + dy * dy + dz * dz + dw * dw;
        }
    }
    #pragma unroll
    for (int s = 1; s < 32; s <<= 1) ls += __shfl_xor(ls, s, 32);
    if (lane == 0) sLoss[wave] = ls;
    __syncthreads();
    if (tid == 0) {
        float s = 0.f;
        #pragma unroll
        for (int w = 0; w < THREADS / 32; ++w) s += sLoss[w];
        wsPartial[blockIdx.x] = s;          // deterministic per-block partial
    }
}

__global__ void vq_finish_kernel(const float* __restrict__ wsPartial,
                                 float* __restrict__ outLoss)
{
    if (blockIdx.x == 0 && threadIdx.x == 0) {
        float s = 0.f;
        for (int i = 0; i < N_WG; ++i) s += wsPartial[i];   // fixed order
        *outLoss = s / (float)(N_ROWS * DIMV);
    }
}

extern "C" void kernel_launch(void* const* d_in, const int* in_sizes, int n_in,
                              void* d_out, int out_size, void* d_ws, size_t ws_size,
                              hipStream_t stream)
{
    (void)in_sizes; (void)n_in; (void)out_size; (void)ws_size;
    const float* inputs = (const float*)d_in[0];   // [32,1024,32] f32
    const float* codes  = (const float*)d_in[1];   // [8192,32] f32 (pre-normalized)
    float* out      = (float*)d_out;
    float* outQuant = out;                          // [32768*32]
    float* outLoss  = out + N_ROWS * DIMV;          // [1]
    float* outIdx   = out + N_ROWS * DIMV + 1;      // [32768] as float
    unsigned int* wsCodes = (unsigned int*)d_ws;    // 1 MB bf16 hi/lo codebook
    float* wsPartial = (float*)((char*)d_ws + (size_t)WS_CODE_U32S * 4);

    vq_prep_kernel<<<N_CODES / 256, 256, 0, stream>>>(codes, wsCodes);
    vq_main_kernel<<<N_WG, THREADS, 0, stream>>>(inputs, codes, wsCodes,
                                                 outQuant, outIdx, wsPartial);
    vq_finish_kernel<<<1, 32, 0, stream>>>(wsPartial, outLoss);
}
